// GridRNNNaive_79783312490693
// MI455X (gfx1250) — compile-verified, matrix operand order checked by reading, not run on hardware
//
#include <hip/hip_runtime.h>
#include <math.h>

// ---------------------------------------------------------------------------
// GridRNN on gfx1250 (MI455X).
//   - All matmuls: v_wmma_f32_16x16x32_f16 (fp32 accumulate, f16 operands)
//   - Recurrences run as persistent kernels: carry ping-pongs in LDS,
//     hidden-weight B-fragments pinned in VGPRs for all 96 steps.
//   - Input projections (incl. biases) hoisted into parallel WMMA GEMMs.
//   9 kernel launches total.
// ---------------------------------------------------------------------------

typedef __attribute__((ext_vector_type(16))) _Float16 v16h;
typedef __attribute__((ext_vector_type(8)))  _Float16 v8h;
typedef __attribute__((ext_vector_type(8)))  float    v8f;
typedef __attribute__((ext_vector_type(4)))  float    v4f;

#define HD   128              // hidden dim
#define NB   8                // batch
#define NQ   96               // S == T == 96
#define ROWS 768              // NB*NQ sequences per scan

// ---- A fragment (16x32 f16, MxK) from a row-major f16 row of length 128 ----
// Lane layout (ISA 7.12.2): lane group g=lane>>4; halves 0..7 hold
// K = kt*32 + g*8 + {0..7}; halves 8..15 hold K = kt*32 + 16 + g*8 + {0..7}.
__device__ __forceinline__ v16h load_a_f16(const _Float16* row, int kt, int g) {
  union { v16h v; v8h h[2]; } a;
  a.h[0] = *(const v8h*)(row + kt * 32 + g * 8);
  a.h[1] = *(const v8h*)(row + kt * 32 + 16 + g * 8);
  return a.v;
}

// Same fragment but converting from an fp32 source row on the fly.
__device__ __forceinline__ v16h load_a_f32(const float* row, int kt, int g) {
  union { v16h v; _Float16 e[16]; } a;
  const float* p0 = row + kt * 32 + g * 8;
  const float* p1 = row + kt * 32 + 16 + g * 8;
  v4f f0 = *(const v4f*)(p0);
  v4f f1 = *(const v4f*)(p0 + 4);
  v4f f2 = *(const v4f*)(p1);
  v4f f3 = *(const v4f*)(p1 + 4);
#pragma unroll
  for (int i = 0; i < 4; ++i) {
    a.e[i]      = (_Float16)f0[i];
    a.e[4 + i]  = (_Float16)f1[i];
    a.e[8 + i]  = (_Float16)f2[i];
    a.e[12 + i] = (_Float16)f3[i];
  }
  return a.v;
}

// ---------------------------------------------------------------------------
// Pack 8 weight matrices (fp32 [128,128], row-major K x N) into WMMA-B f16
// fragment layout: Bpack[mat][kt][nt][lane][16 halves].
// mat = (src<<1)|depth, src: 0=Wx_ih 1=Wx_hh 2=Wy_ih 3=Wy_hh.
// ---------------------------------------------------------------------------
__global__ __launch_bounds__(256) void pack_weights(
    const float* __restrict__ Wx_ih, const float* __restrict__ Wx_hh,
    const float* __restrict__ Wy_ih, const float* __restrict__ Wy_hh,
    _Float16* __restrict__ Bpack) {
  int t    = blockIdx.x * 256 + threadIdx.x;   // 8192 threads total
  int lane = t & 31;
  int nt   = (t >> 5) & 7;
  int kt   = (t >> 8) & 3;
  int mat  = t >> 10;
  const float* srcs[4] = {Wx_ih, Wx_hh, Wy_ih, Wy_hh};
  const float* W = srcs[mat >> 1] + (size_t)(mat & 1) * HD * HD;
  int n = nt * 16 + (lane & 15);
  int g = lane >> 4;
  union { v16h v; _Float16 e[16]; } f;
#pragma unroll
  for (int v = 0; v < 16; ++v) {
    int k = kt * 32 + ((v < 8) ? (g * 8 + v) : (16 + g * 8 + (v - 8)));
    f.e[v] = (_Float16)W[k * HD + n];
  }
  *(v16h*)(Bpack + (size_t)t * 16) = f.v;
}

// ---------------------------------------------------------------------------
// G = A(f32)[768,128] @ Wpack + bias0 + bias1   (depth-0 input projection)
// ---------------------------------------------------------------------------
__global__ __launch_bounds__(256) void proj_f32(
    const float* __restrict__ A, const _Float16* __restrict__ Wpack,
    const float* __restrict__ bias0, const float* __restrict__ bias1,
    float* __restrict__ G) {
  const int lane = threadIdx.x & 31;
  const int w    = threadIdx.x >> 5;
  const int g    = lane >> 4;
  const int m    = lane & 15;
  const int tile = blockIdx.x;
  const float* arow = A + (size_t)(tile * 16 + m) * HD;

  v8f acc = {};
#pragma unroll
  for (int kt = 0; kt < 4; ++kt) {
    v16h a  = load_a_f32(arow, kt, g);
    v16h bm = *(const v16h*)(Wpack + (size_t)((kt * 8 + w) * 32 + lane) * 16);
    acc = __builtin_amdgcn_wmma_f32_16x16x32_f16(false, a, false, bm,
                                                 (short)0, acc, false, false);
  }
  const int col  = w * 16 + m;
  const float bs = bias0[col] + bias1[col];
#pragma unroll
  for (int r = 0; r < 8; ++r) {
    int gr = tile * 16 + r + 8 * g;              // C/D layout: row = r + 8*g
    G[(size_t)gr * HD + col] = acc[r] + bs;
  }
}

// ---------------------------------------------------------------------------
// Depth-1 input projection: G[b,i,j] = Hprev[b,i,(j-1)%96] @ Wpack + biases.
// grid.x = 73728/16 = 4608 tiles (j fastest; tiles never straddle (b,i)).
// ---------------------------------------------------------------------------
__global__ __launch_bounds__(256) void proj_rolled(
    const _Float16* __restrict__ Hprev, const _Float16* __restrict__ Wpack,
    const float* __restrict__ bias0, const float* __restrict__ bias1,
    float* __restrict__ G) {
  const int lane = threadIdx.x & 31;
  const int w    = threadIdx.x >> 5;
  const int g    = lane >> 4;
  const int m    = lane & 15;
  const int tile = blockIdx.x;
  const int gr   = tile * 16 + m;
  const int cj   = gr % NQ;
  const int bi   = gr / NQ;
  const _Float16* arow =
      Hprev + ((size_t)bi * NQ + (cj == 0 ? NQ - 1 : cj - 1)) * HD;

  v8f acc = {};
#pragma unroll
  for (int kt = 0; kt < 4; ++kt) {
    v16h a  = load_a_f16(arow, kt, g);
    v16h bm = *(const v16h*)(Wpack + (size_t)((kt * 8 + w) * 32 + lane) * 16);
    acc = __builtin_amdgcn_wmma_f32_16x16x32_f16(false, a, false, bm,
                                                 (short)0, acc, false, false);
  }
  const int col  = w * 16 + m;
  const float bs = bias0[col] + bias1[col];
#pragma unroll
  for (int r = 0; r < 8; ++r) {
    int gr2 = tile * 16 + r + 8 * g;
    G[(size_t)gr2 * HD + col] = acc[r] + bs;
  }
}

// ---------------------------------------------------------------------------
// Persistent y-scan: 96 steps over j; block owns 16 rows (fixed b); carry
// ping-pongs in LDS; B-fragments of W_hh pinned in VGPRs for all steps.
//   h_new = tanh( carry @ W_hh + G )
// ---------------------------------------------------------------------------
__global__ __launch_bounds__(256) void scan_y_persist(
    const _Float16* __restrict__ Wpack, const float* __restrict__ G,
    float* __restrict__ outp, _Float16* __restrict__ h16, int write_h16,
    int g_bs, int g_qs, int g_ss,
    long o_bs, long o_qs, long o_ss,
    int h_qs, int h_ss) {
  __shared__ _Float16 lds[2][16 * HD];           // 2 x 4KB carry ping-pong
  const int lane = threadIdx.x & 31;
  const int w    = threadIdx.x >> 5;
  const int g    = lane >> 4;
  const int m    = lane & 15;
  const int tile = blockIdx.x;

  v16h bfrag[4];
#pragma unroll
  for (int kt = 0; kt < 4; ++kt)
    bfrag[kt] = *(const v16h*)(Wpack + (size_t)((kt * 8 + w) * 32 + lane) * 16);

  {  // zero initial carry (16*128 halves / 256 threads = 8 each)
    v8h z = {};
    *(v8h*)(&lds[0][threadIdx.x * 8]) = z;
  }
  __syncthreads();

  const int col = w * 16 + m;
  for (int step = 0; step < NQ; ++step) {
    const int p = step & 1;
    const _Float16* crow = &lds[p][m * HD];
    v8f acc = {};
#pragma unroll
    for (int kt = 0; kt < 4; ++kt) {
      v16h a = load_a_f16(crow, kt, g);
      acc = __builtin_amdgcn_wmma_f32_16x16x32_f16(false, a, false, bfrag[kt],
                                                   (short)0, acc, false, false);
    }
#pragma unroll
    for (int r = 0; r < 8; ++r) {
      int mr  = r + 8 * g;
      int gr2 = tile * 16 + mr;
      int b2  = gr2 / NQ;
      int q2  = gr2 % NQ;
      float gv = G[(size_t)b2 * g_bs + (size_t)q2 * g_qs +
                   (size_t)step * g_ss + col];
      float v  = tanhf(acc[r] + gv);
      outp[(size_t)b2 * o_bs + (size_t)q2 * o_qs + (size_t)step * o_ss + col] = v;
      _Float16 hv = (_Float16)v;
      lds[1 - p][mr * HD + col] = hv;
      if (write_h16)
        h16[(size_t)b2 * (NQ * NQ * HD) + (size_t)q2 * h_qs +
            (size_t)step * h_ss + col] = hv;
    }
    __syncthreads();   // publish lds[1-p] before next step reads it
  }
}

// ---------------------------------------------------------------------------
// Persistent x-scan: 96 steps over i; one block per b owns all 96 j-rows so
// the carry roll  st[j] = carry[(j-1)%96]  is a pure LDS ping-pong read.
// ---------------------------------------------------------------------------
__global__ __launch_bounds__(256) void scan_x_persist(
    const _Float16* __restrict__ Wpack, const float* __restrict__ G,
    float* __restrict__ outp, _Float16* __restrict__ h16, int write_h16,
    int g_bs, int g_qs, int g_ss,
    long o_bs, long o_qs, long o_ss,
    int h_qs, int h_ss) {
  __shared__ _Float16 lds[2][NQ * HD];           // 2 x 24KB carry ping-pong
  const int lane = threadIdx.x & 31;
  const int w    = threadIdx.x >> 5;
  const int g    = lane >> 4;
  const int m    = lane & 15;
  const int b    = blockIdx.x;

  v16h bfrag[4];
#pragma unroll
  for (int kt = 0; kt < 4; ++kt)
    bfrag[kt] = *(const v16h*)(Wpack + (size_t)((kt * 8 + w) * 32 + lane) * 16);

  {  // zero initial carry (96*128 halves / 256 threads = 48 each)
    v8h z = {};
#pragma unroll
    for (int zz = 0; zz < 6; ++zz)
      *(v8h*)(&lds[0][zz * 2048 + threadIdx.x * 8]) = z;
  }
  __syncthreads();

  const int col = w * 16 + m;
  for (int step = 0; step < NQ; ++step) {
    const int p = step & 1;
    for (int rt = 0; rt < 6; ++rt) {
      const int j  = rt * 16 + m;
      const int ja = (j == 0) ? NQ - 1 : j - 1;   // rolled carry row
      const _Float16* crow = &lds[p][ja * HD];
      v8f acc = {};
#pragma unroll
      for (int kt = 0; kt < 4; ++kt) {
        v16h a = load_a_f16(crow, kt, g);
        acc = __builtin_amdgcn_wmma_f32_16x16x32_f16(
            false, a, false, bfrag[kt], (short)0, acc, false, false);
      }
#pragma unroll
      for (int r = 0; r < 8; ++r) {
        int j2 = rt * 16 + r + 8 * g;
        float gv = G[(size_t)b * g_bs + (size_t)j2 * g_qs +
                     (size_t)step * g_ss + col];
        float v  = tanhf(acc[r] + gv);
        outp[(size_t)b * o_bs + (size_t)j2 * o_qs + (size_t)step * o_ss + col] = v;
        _Float16 hv = (_Float16)v;
        lds[1 - p][j2 * HD + col] = hv;
        if (write_h16)
          h16[(size_t)b * (NQ * NQ * HD) + (size_t)j2 * h_qs +
              (size_t)step * h_ss + col] = hv;
      }
    }
    __syncthreads();   // publish lds[1-p] before next step reads it
  }
}

// ---------------------------------------------------------------------------
extern "C" void kernel_launch(void* const* d_in, const int* in_sizes, int n_in,
                              void* d_out, int out_size, void* d_ws,
                              size_t ws_size, hipStream_t stream) {
  (void)in_sizes; (void)n_in; (void)out_size; (void)ws_size;
  const float* x     = (const float*)d_in[0];
  const float* y     = (const float*)d_in[1];
  const float* Wx_ih = (const float*)d_in[2];
  const float* Wx_hh = (const float*)d_in[3];
  const float* bx_ih = (const float*)d_in[4];
  const float* bx_hh = (const float*)d_in[5];
  const float* Wy_ih = (const float*)d_in[6];
  const float* Wy_hh = (const float*)d_in[7];
  const float* by_ih = (const float*)d_in[8];
  const float* by_hh = (const float*)d_in[9];
  float* out = (float*)d_out;

  // ---- workspace carve-up (256B aligned) ----
  char* ws = (char*)d_ws;
  size_t off = 0;
  auto carve = [&](size_t bytes) {
    void* p = ws + off;
    off = (off + bytes + 255) & ~(size_t)255;
    return p;
  };
  _Float16* Bpack = (_Float16*)carve((size_t)8 * 4 * 8 * 32 * 16 * 2); // 256KB
  float*    Gx0   = (float*)carve((size_t)ROWS * HD * 4);
  float*    Gy0   = (float*)carve((size_t)ROWS * HD * 4);
  float*    Gx1   = (float*)carve((size_t)NB * NQ * NQ * HD * 4);      // 37.7MB
  float*    Gy1   = (float*)carve((size_t)NB * NQ * NQ * HD * 4);
  _Float16* hx16  = (_Float16*)carve((size_t)NB * NQ * NQ * HD * 2);   // 18.9MB
  _Float16* hy16  = (_Float16*)carve((size_t)NB * NQ * NQ * HD * 2);

  auto matp = [&](int i) { return Bpack + (size_t)i * 4 * 8 * 32 * 16; };

  // output strides (elements, fp32): [B, D, S, T, 2, H]
  const long OB = 2L * NQ * NQ * 2 * HD;  // per-b
  const long OD = (long)NQ * NQ * 2 * HD; // per-depth
  const long Oi = (long)NQ * 2 * HD;      // per-i
  const long Oj = 2L * HD;                // per-j

  // ---- pack weights, precompute depth-0 projections ----
  pack_weights<<<32, 256, 0, stream>>>(Wx_ih, Wx_hh, Wy_ih, Wy_hh, Bpack);
  proj_f32<<<ROWS / 16, 256, 0, stream>>>(x, matp(0), bx_ih, bx_hh, Gx0);
  proj_f32<<<ROWS / 16, 256, 0, stream>>>(y, matp(4), by_ih, by_hh, Gy0);

  for (int d = 0; d < 2; ++d) {
    // y-scan: rows = (b,i), step = j, no roll
    scan_y_persist<<<ROWS / 16, 256, 0, stream>>>(
        matp(6 + d), d ? Gy1 : Gy0, out + d * OD + HD /*h_y*/, hy16,
        /*write_h16=*/d == 0 ? 1 : 0,
        /*g_bs*/ d ? NQ * NQ * HD : NQ * HD,
        /*g_qs*/ d ? NQ * HD : 0,
        /*g_ss*/ HD,
        OB, /*o_qs(i)*/ Oi, /*o_ss(j)*/ Oj,
        /*h_qs(i)*/ NQ * HD, /*h_ss(j)*/ HD);

    // x-scan: rows = (b,j), step = i, carry rolled along j
    scan_x_persist<<<NB, 256, 0, stream>>>(
        matp(2 + d), d ? Gx1 : Gx0, out + d * OD /*h_x*/, hx16,
        /*write_h16=*/d == 0 ? 1 : 0,
        /*g_bs*/ d ? NQ * NQ * HD : NQ * HD,
        /*g_qs*/ d ? HD : 0,
        /*g_ss*/ d ? NQ * HD : HD,
        OB, /*o_qs(j)*/ Oj, /*o_ss(i)*/ Oi,
        /*h_qs(j)*/ HD, /*h_ss(i)*/ NQ * HD);

    // depth-1 input projections from rolled depth-0 hiddens
    if (d == 0) {
      proj_rolled<<<NB * NQ * NQ / 16, 256, 0, stream>>>(
          hy16, matp(5), by_ih + HD, by_hh + HD, Gy1);
      proj_rolled<<<NB * NQ * NQ / 16, 256, 0, stream>>>(
          hx16, matp(1), bx_ih + HD, bx_hh + HD, Gx1);
    }
  }
}
